// HGNNConv_49692771615016
// MI455X (gfx1250) — matrix operand order repaired and checked by reading, not compiled, not run on hardware
//
#include <hip/hip_runtime.h>
#include <math.h>

#define NN 50000
#define EE 800000
#define RR 4
#define HH 8
#define NEG_SLOPE 0.2f

typedef __attribute__((ext_vector_type(16))) _Float16 v16h;
typedef __attribute__((ext_vector_type(8)))  _Float16 v8h;
typedef __attribute__((ext_vector_type(8)))  float    v8f;

__device__ __forceinline__ float leaky(float x) { return x > 0.f ? x : NEG_SLOPE * x; }
__device__ __forceinline__ float gelu_exact(float x) {
  return 0.5f * x * (1.f + erff(x * 0.70710678118654752440f));
}
__device__ __forceinline__ float dot4(float4 a, float4 b) {
  return a.x * b.x + a.y * b.y + a.z * b.z + a.w * b.w;
}
// monotonic float <-> uint mapping for atomicMax on floats
__device__ __forceinline__ unsigned f2ord(float f) {
  unsigned u = __float_as_uint(f);
  return (u & 0x80000000u) ? ~u : (u | 0x80000000u);
}
__device__ __forceinline__ float ord2f(unsigned u) {
  return (u & 0x80000000u) ? __uint_as_float(u & 0x7fffffffu) : __uint_as_float(~u);
}

// ---------------------------------------------------------------------------
// GEMM: C[nrows,128] = act(A[nrows,128] (+bias)) @ W[128,128], f16 WMMA, f32 acc
// ---------------------------------------------------------------------------
#define GWAVES 4
#define GROWS (GWAVES * 16)
#define SW_STRIDE 144   // halfs; 288B row stride -> conflict-light b128 reads
#define SA_STRIDE 136   // halfs; 272B row stride

template <bool ACT>
__global__ __launch_bounds__(GWAVES * 32) void gemm128(
    const float* __restrict__ A, const float* __restrict__ W,
    const float* __restrict__ bias, float* __restrict__ C, int nrows) {
  __shared__ __attribute__((aligned(32))) _Float16 sW[128 * SW_STRIDE];
  __shared__ __attribute__((aligned(32))) _Float16 sA[GWAVES * 16 * SA_STRIDE];

  const int tid = threadIdx.x;
  const int wave = tid >> 5;
  const int lane = tid & 31;
  const int lo = lane & 15;
  const int hi = lane >> 4;
  const int rowbase = blockIdx.x * GROWS + wave * 16;
  const bool full = (rowbase + 16 <= nrows);  // all rows of this wave valid

  // Stage W^T into LDS as f16: sW[n][k] = W[k][n]
  for (int i = tid; i < 128 * 128; i += GWAVES * 32) {
    int k = i >> 7, n = i & 127;
    sW[n * SW_STRIDE + k] = (_Float16)W[i];
  }

  // Stage this wave's 16 rows of A (f32 -> f16, optional bias + gelu)
  {
    const int c = lane * 4;
    float bx = 0.f, by = 0.f, bz = 0.f, bw = 0.f;
    if (ACT) { bx = bias[c]; by = bias[c + 1]; bz = bias[c + 2]; bw = bias[c + 3]; }
    const float* arow0 = A + (size_t)rowbase * 128 + c;
    _Float16* drow0 = &sA[(wave * 16) * SA_STRIDE + c];
    if (full) {
#pragma unroll
      for (int r = 0; r < 16; ++r) {
        float4 v = *(const float4*)(arow0 + (size_t)r * 128);
        if (ACT) {
          v.x = gelu_exact(v.x + bx); v.y = gelu_exact(v.y + by);
          v.z = gelu_exact(v.z + bz); v.w = gelu_exact(v.w + bw);
        }
        _Float16* dst = drow0 + r * SA_STRIDE;
        dst[0] = (_Float16)v.x; dst[1] = (_Float16)v.y;
        dst[2] = (_Float16)v.z; dst[3] = (_Float16)v.w;
      }
    } else {
      for (int r = 0; r < 16; ++r) {
        float4 v = make_float4(0.f, 0.f, 0.f, 0.f);
        if (rowbase + r < nrows) v = *(const float4*)(arow0 + (size_t)r * 128);
        if (ACT) {
          v.x = gelu_exact(v.x + bx); v.y = gelu_exact(v.y + by);
          v.z = gelu_exact(v.z + bz); v.w = gelu_exact(v.w + bw);
        }
        _Float16* dst = drow0 + r * SA_STRIDE;
        dst[0] = (_Float16)v.x; dst[1] = (_Float16)v.y;
        dst[2] = (_Float16)v.z; dst[3] = (_Float16)v.w;
      }
    }
  }
  __syncthreads();

  // A fragments for the 4 K-steps (contiguous K order, same as B: lane row = lo,
  // element e -> K = ks*32 + hi*16 + e)
  v16h afrag[4];
  const _Float16* arow = &sA[(wave * 16 + lo) * SA_STRIDE];
#pragma unroll
  for (int ks = 0; ks < 4; ++ks) {
    const _Float16* ap = arow + ks * 32 + hi * 16;
    v8h a0 = *(const v8h*)(ap);
    v8h a1 = *(const v8h*)(ap + 8);
    afrag[ks] = __builtin_shufflevector(a0, a1, 0, 1, 2, 3, 4, 5, 6, 7, 8, 9,
                                        10, 11, 12, 13, 14, 15);
  }

  // per-lane output base: row = rowbase + hi*8 (+r), col = lo (+nt*16)
  float* crow = C + (size_t)(rowbase + hi * 8) * 128 + lo;

#pragma unroll
  for (int nt = 0; nt < 8; ++nt) {
    v8f acc = {0.f, 0.f, 0.f, 0.f, 0.f, 0.f, 0.f, 0.f};
    const _Float16* bcol = &sW[(nt * 16 + lo) * SW_STRIDE];
#pragma unroll
    for (int ks = 0; ks < 4; ++ks) {
      // B-layout: lane column = lo, element e -> K = ks*32 + hi*16 + e
      v16h b = *(const v16h*)(bcol + ks * 32 + hi * 16);
      acc = __builtin_amdgcn_wmma_f32_16x16x32_f16(false, afrag[ks], false, b,
                                                   (short)0, acc, false, false);
    }
    if (full) {
#pragma unroll
      for (int r = 0; r < 8; ++r) crow[r * 128 + nt * 16] = acc[r];
    } else {
      for (int r = 0; r < 8; ++r)
        if (rowbase + hi * 8 + r < nrows) crow[r * 128 + nt * 16] = acc[r];
    }
  }
}

// ---------------------------------------------------------------------------
// es/ed per (r, n, h)
// ---------------------------------------------------------------------------
__global__ void attn_coef(const float* __restrict__ x,
                          const float* __restrict__ a_src,
                          const float* __restrict__ a_dst,
                          float* __restrict__ es, float* __restrict__ ed) {
  int idx = blockIdx.x * blockDim.x + threadIdx.x;
  if (idx >= RR * NN * HH) return;
  int h = idx & 7;
  int n = (idx >> 3) % NN;
  int r = idx / (NN * HH);
  const float4* xp = (const float4*)(x + ((size_t)r * NN + n) * 128 + h * 16);
  const float4* as = (const float4*)(a_src + (r * HH + h) * 16);
  const float4* ad = (const float4*)(a_dst + (r * HH + h) * 16);
  float s = 0.f, d = 0.f;
#pragma unroll
  for (int j = 0; j < 4; ++j) {
    float4 xv = xp[j];
    s += dot4(xv, as[j]);
    d += dot4(xv, ad[j]);
  }
  es[idx] = s;
  ed[idx] = d;
}

// ---------------------------------------------------------------------------
// init: mx = ord(-1e30), den = 0, xagg = 0
// ---------------------------------------------------------------------------
__global__ void init_ws(unsigned* __restrict__ mx, float* __restrict__ den,
                        float* __restrict__ xagg) {
  int i = blockIdx.x * blockDim.x + threadIdx.x;
  if (i < RR * NN * HH) {
    mx[i] = f2ord(-1e30f);
    den[i] = 0.f;
  }
  if (i < RR * NN * 128) xagg[i] = 0.f;
}

// ---------------------------------------------------------------------------
// edge pass 1: segment max of leaky(es[src]+ed[dst]) into mx (ordered-uint)
// ---------------------------------------------------------------------------
__global__ void edge_max(const int* __restrict__ ei, const int* __restrict__ et,
                         const float* __restrict__ es,
                         const float* __restrict__ ed,
                         unsigned* __restrict__ mx) {
  int idx = blockIdx.x * blockDim.x + threadIdx.x;
  if (idx >= EE * HH) return;
  int h = idx & 7;
  int e = idx >> 3;
  int s = ei[e], d = ei[EE + e], r = et[e];
  float lg = leaky(es[((size_t)r * NN + s) * HH + h] +
                   ed[((size_t)r * NN + d) * HH + h]);
  atomicMax(&mx[((size_t)r * NN + d) * HH + h], f2ord(lg));
}

// ---------------------------------------------------------------------------
// edge pass 2: den += exp(logit - mx[dst])
// ---------------------------------------------------------------------------
__global__ void edge_den(const int* __restrict__ ei, const int* __restrict__ et,
                         const float* __restrict__ es,
                         const float* __restrict__ ed,
                         const unsigned* __restrict__ mx,
                         float* __restrict__ den) {
  int idx = blockIdx.x * blockDim.x + threadIdx.x;
  if (idx >= EE * HH) return;
  int h = idx & 7;
  int e = idx >> 3;
  int s = ei[e], d = ei[EE + e], r = et[e];
  size_t o = ((size_t)r * NN + d) * HH + h;
  float lg = leaky(es[((size_t)r * NN + s) * HH + h] + ed[o]);
  atomicAdd(&den[o], __expf(lg - ord2f(mx[o])));
}

// ---------------------------------------------------------------------------
// edge pass 3: xagg[dst] += alpha * x[src]
// ---------------------------------------------------------------------------
__global__ void edge_scatter(const int* __restrict__ ei,
                             const int* __restrict__ et,
                             const float* __restrict__ es,
                             const float* __restrict__ ed,
                             const unsigned* __restrict__ mx,
                             const float* __restrict__ den,
                             const float* __restrict__ x,
                             float* __restrict__ xagg) {
  int idx = blockIdx.x * blockDim.x + threadIdx.x;
  if (idx >= EE * HH) return;
  int h = idx & 7;
  int e = idx >> 3;
  int s = ei[e], d = ei[EE + e], r = et[e];
  size_t o = ((size_t)r * NN + d) * HH + h;
  float lg = leaky(es[((size_t)r * NN + s) * HH + h] + ed[o]);
  float dn = den[o];
  float alpha = __expf(lg - ord2f(mx[o])) / (dn > 0.f ? dn : 1.f);
  const float4* xs = (const float4*)(x + ((size_t)r * NN + s) * 128 + h * 16);
  float* xa = xagg + ((size_t)r * NN + d) * 128 + h * 16;
#pragma unroll
  for (int j = 0; j < 4; ++j) {
    float4 v = xs[j];
    atomicAdd(xa + 4 * j + 0, alpha * v.x);
    atomicAdd(xa + 4 * j + 1, alpha * v.y);
    atomicAdd(xa + 4 * j + 2, alpha * v.z);
    atomicAdd(xa + 4 * j + 3, alpha * v.w);
  }
}

// ---------------------------------------------------------------------------
// lang conv: per (n,h) softmax over {self} U {rel outputs where node is target}
// ---------------------------------------------------------------------------
__global__ void lang_conv(const float* __restrict__ S,
                          const float* __restrict__ hx,  // [R][N][128]
                          const float* __restrict__ den, // [R][N][H]
                          const float* __restrict__ as_l,
                          const float* __restrict__ ad_l,
                          const float* __restrict__ bias_l,
                          float* __restrict__ out) {
  int idx = blockIdx.x * blockDim.x + threadIdx.x;
  if (idx >= NN * HH) return;
  int h = idx & 7;
  int n = idx >> 3;

  float4 ent[RR + 1][4];
  const float4* sp = (const float4*)(S + (size_t)n * 128 + h * 16);
#pragma unroll
  for (int j = 0; j < 4; ++j) ent[0][j] = sp[j];
#pragma unroll
  for (int k = 1; k <= RR; ++k) {
    const float4* hp =
        (const float4*)(hx + ((size_t)(k - 1) * NN + n) * 128 + h * 16);
#pragma unroll
    for (int j = 0; j < 4; ++j) ent[k][j] = hp[j];
  }

  const float4* asv = (const float4*)(as_l + h * 16);
  const float4* adv = (const float4*)(ad_l + h * 16);
  float ld = 0.f;
#pragma unroll
  for (int j = 0; j < 4; ++j) ld += dot4(ent[0][j], adv[j]);

  float logit[RR + 1];
#pragma unroll
  for (int k = 0; k <= RR; ++k) {
    float ls = 0.f;
#pragma unroll
    for (int j = 0; j < 4; ++j) ls += dot4(ent[k][j], asv[j]);
    bool valid = (k == 0) || (den[((size_t)(k - 1) * NN + n) * HH + 0] > 0.f);
    logit[k] = valid ? leaky(ls + ld) : -1e30f;
  }
  float m = logit[0];
#pragma unroll
  for (int k = 1; k <= RR; ++k) m = fmaxf(m, logit[k]);
  float p[RR + 1], sum = 0.f;
#pragma unroll
  for (int k = 0; k <= RR; ++k) {
    p[k] = __expf(logit[k] - m);
    sum += p[k];
  }
  float inv = 1.f / sum;

  const float4* bp = (const float4*)(bias_l + h * 16);
  float4* op = (float4*)(out + (size_t)n * 128 + h * 16);
#pragma unroll
  for (int j = 0; j < 4; ++j) {
    float4 o = bp[j];
#pragma unroll
    for (int k = 0; k <= RR; ++k) {
      float a = p[k] * inv;
      o.x += a * ent[k][j].x;
      o.y += a * ent[k][j].y;
      o.z += a * ent[k][j].z;
      o.w += a * ent[k][j].w;
    }
    op[j] = o;
  }
}

// ---------------------------------------------------------------------------
extern "C" void kernel_launch(void* const* d_in, const int* in_sizes, int n_in,
                              void* d_out, int out_size, void* d_ws,
                              size_t ws_size, hipStream_t stream) {
  (void)in_sizes; (void)n_in; (void)out_size; (void)ws_size;

  const float* node_inp     = (const float*)d_in[0];
  const int*   edge_index   = (const int*)d_in[1];
  const int*   edge_type    = (const int*)d_in[2];
  const float* W_rel        = (const float*)d_in[3];
  const float* att_src_rel  = (const float*)d_in[4];
  const float* att_dst_rel  = (const float*)d_in[5];
  const float* bias_rel     = (const float*)d_in[6];
  const float* W_self       = (const float*)d_in[7];
  const float* W_cross      = (const float*)d_in[8];
  const float* att_src_lang = (const float*)d_in[9];
  const float* att_dst_lang = (const float*)d_in[10];
  const float* bias_lang    = (const float*)d_in[11];
  float* out = (float*)d_out;

  // workspace carve-up
  char* ws = (char*)d_ws;
  size_t off = 0;
  float* x = (float*)(ws + off);        off += (size_t)RR * NN * 128 * 4; // x_r, later hcross
  float* S = (float*)(ws + off);        off += (size_t)NN * 128 * 4;
  float* es = (float*)(ws + off);       off += (size_t)RR * NN * HH * 4;
  float* ed = (float*)(ws + off);       off += (size_t)RR * NN * HH * 4;
  unsigned* mx = (unsigned*)(ws + off); off += (size_t)RR * NN * HH * 4;
  float* den = (float*)(ws + off);      off += (size_t)RR * NN * HH * 4;
  float* xagg = (float*)(ws + off);     off += (size_t)RR * NN * 128 * 4;

  const int gemm_blocks = (NN + GROWS - 1) / GROWS;

  // 1) x_r = node_inp @ W_rel[r], S = node_inp @ W_self (WMMA)
  for (int r = 0; r < RR; ++r)
    gemm128<false><<<gemm_blocks, GWAVES * 32, 0, stream>>>(
        node_inp, W_rel + (size_t)r * 128 * 128, nullptr,
        x + (size_t)r * NN * 128, NN);
  gemm128<false><<<gemm_blocks, GWAVES * 32, 0, stream>>>(node_inp, W_self,
                                                          nullptr, S, NN);

  // 2) attention coefficients
  attn_coef<<<(RR * NN * HH + 255) / 256, 256, 0, stream>>>(
      x, att_src_rel, att_dst_rel, es, ed);

  // 3) init accumulators
  init_ws<<<(RR * NN * 128 + 255) / 256, 256, 0, stream>>>(mx, den, xagg);

  // 4) edge softmax aggregation (3 passes)
  const int eblocks = (EE * HH + 255) / 256;
  edge_max<<<eblocks, 256, 0, stream>>>(edge_index, edge_type, es, ed, mx);
  edge_den<<<eblocks, 256, 0, stream>>>(edge_index, edge_type, es, ed, mx, den);
  edge_scatter<<<eblocks, 256, 0, stream>>>(edge_index, edge_type, es, ed, mx,
                                            den, x, xagg);

  // 5) h = gelu(xagg + bias_rel) @ W_cross  (overwrites x_r buffer)
  for (int r = 0; r < RR; ++r)
    gemm128<true><<<gemm_blocks, GWAVES * 32, 0, stream>>>(
        xagg + (size_t)r * NN * 128, W_cross, bias_rel + r * 128,
        x + (size_t)r * NN * 128, NN);

  // 6) lang conv + bias
  lang_conv<<<(NN * HH + 255) / 256, 256, 0, stream>>>(
      S, x, den, att_src_lang, att_dst_lang, bias_lang, out);
}
